// GAEModel_36867999269273
// MI455X (gfx1250) — matrix operand order repaired and checked by reading, not compiled, not run on hardware
//
#include <hip/hip_runtime.h>
#include <hip/hip_bf16.h>

// ---------------------------------------------------------------------------
// GAE forward for gfx1250 (MI455X).  Heavy GEMMs use v_wmma_f32_16x16x32_f16;
// edge aggregation uses global_atomic_add_f32 (L2-resident: h is 51 MB < 192 MB L2).
// ---------------------------------------------------------------------------

typedef __attribute__((ext_vector_type(16))) _Float16 v16h;
typedef __attribute__((ext_vector_type(8)))  float    v8f;

#define HID 256
#define NLAYERS 3

union AF { v16h v; uint4 q[2]; };

// ---------------- WMMA GEMM: out[M x 256] = A[M x K] * Bt  (+bias, relu) ----
// A row-major f16.  Bt is [256][K] f16 (column n of the logical B, k-contiguous).
// One wave (32 threads) per 16-row strip; 16 column tiles of 16.
// Fragment layouts per CDNA5 ISA 7.12.2:
//   A 16x32 f16 : lane L -> M=L%16 ; halves i=0..7 -> K=kb+ (L<16?0:8)+i,
//                 i=8..15 -> K=kb+16+(L<16?0:8)+(i-8)
//   B 32x16 f16 : lane L -> N=L%16 ; halves i -> K as in A above (same table order)
//   C/D 16x16 f32: lane L, reg r -> M=r+(L<16?0:8), N=L%16
// All configuration is compile-time so the epilogue is branch-free.
template<int KSTEPS, bool HAS_BIAS, bool RELU, bool OUT_H, bool OUT_F>
__global__ void wmma_gemm(const _Float16* A, const _Float16* Bt,
                          const float* __restrict__ bias, int K,
                          _Float16* outH, float* outF)
{
    const int lane = threadIdx.x & 31;
    const int m    = lane & 15;
    const int hi   = lane >> 4;
    const long long row0 = (long long)blockIdx.x * 16;

    AF a[KSTEPS];
    const _Float16* arow = A + (row0 + m) * (long long)K;
#pragma unroll
    for (int j = 0; j < KSTEPS; ++j) {
        const int kb = j * 32;
        a[j].q[0] = *(const uint4*)(arow + kb +      hi * 8);
        a[j].q[1] = *(const uint4*)(arow + kb + 16 + hi * 8);
    }

    for (int t = 0; t < 16; ++t) {
        const int n = t * 16 + m;
        const _Float16* bcol = Bt + (long long)n * K;
        v8f c = {};
#pragma unroll
        for (int j = 0; j < KSTEPS; ++j) {
            AF b;
            const int kb = j * 32 + hi * 8;
            b.q[0] = *(const uint4*)(bcol + kb);
            b.q[1] = *(const uint4*)(bcol + kb + 16);
            c = __builtin_amdgcn_wmma_f32_16x16x32_f16(
                    false, a[j].v, false, b.v, (short)0, c, false, false);
        }
        float bv = 0.0f;
        if (HAS_BIAS) bv = bias[n];
#pragma unroll
        for (int r = 0; r < 8; ++r) {
            const long long row = row0 + r + hi * 8;
            float v = c[r] + bv;
            if (RELU) v = fmaxf(v, 0.0f);
            const long long idx = row * HID + n;
            if (OUT_F) outF[idx] = v;
            if (OUT_H) outH[idx] = (_Float16)v;
        }
    }
}

// ---------------- helpers ---------------------------------------------------

// Wt[n*K + k] = (f16) W[k*Nout + n]   (convert + transpose)
__global__ void w_transpose_f16(const float* __restrict__ W, _Float16* __restrict__ Wt,
                                int K, int Nout)
{
    int idx = blockIdx.x * 256 + threadIdx.x;
    if (idx >= K * Nout) return;
    int k = idx / Nout, n = idx - k * Nout;
    Wt[(long long)n * K + k] = (_Float16)W[idx];
}

__global__ void f32_to_f16(const float* __restrict__ in, _Float16* __restrict__ out,
                           long long n)
{
    long long i = (long long)blockIdx.x * 256 + threadIdx.x;
    if (i < n) out[i] = (_Float16)in[i];
}

__global__ void deg_init(float* deg, int N)
{
    int i = blockIdx.x * 256 + threadIdx.x;
    if (i < N) deg[i] = 1.0f;               // self-loop contribution
}

__global__ void deg_accum(const int* __restrict__ dst, float* deg, int E)
{
    int i = blockIdx.x * 256 + threadIdx.x;
    if (i < E)
        __hip_atomic_fetch_add(&deg[dst[i]], 1.0f,
                               __ATOMIC_RELAXED, __HIP_MEMORY_SCOPE_AGENT);
}

__global__ void deg_rsqrt(float* deg, int N)
{
    int i = blockIdx.x * 256 + threadIdx.x;
    if (i < N) deg[i] = rsqrtf(deg[i]);      // deg >= 1 always (self loops)
}

__global__ void acc_init(float* __restrict__ acc, const float* __restrict__ b,
                         long long total)
{
    long long i = (long long)blockIdx.x * 256 + threadIdx.x;
    if (i < total) acc[i] = b[i & (HID - 1)];
}

// msg scatter: acc[dst] += hw[src] * dinv[src]*dinv[dst]  over E real + N self edges.
// One lane handles 8 features of one edge (32 lanes cover HID=256).
__global__ void scatter_edges(const float* __restrict__ hw, float* __restrict__ acc,
                              const int* __restrict__ src, const int* __restrict__ dst,
                              const float* __restrict__ dinv, int E, int N)
{
    long long gid = (long long)blockIdx.x * 256 + threadIdx.x;
    int e = (int)(gid >> 5);
    if (e >= E + N) return;
    int lane = (int)(gid & 31);
    int s, d;
    if (e < E) { s = src[e]; d = dst[e]; } else { s = d = e - E; }
    float nrm = dinv[s] * dinv[d];
    const float4* p = (const float4*)(hw + (long long)s * HID + lane * 8);
    float4 a0 = p[0], a1 = p[1];
    float* q = acc + (long long)d * HID + lane * 8;
    __hip_atomic_fetch_add(q + 0, a0.x * nrm, __ATOMIC_RELAXED, __HIP_MEMORY_SCOPE_AGENT);
    __hip_atomic_fetch_add(q + 1, a0.y * nrm, __ATOMIC_RELAXED, __HIP_MEMORY_SCOPE_AGENT);
    __hip_atomic_fetch_add(q + 2, a0.z * nrm, __ATOMIC_RELAXED, __HIP_MEMORY_SCOPE_AGENT);
    __hip_atomic_fetch_add(q + 3, a0.w * nrm, __ATOMIC_RELAXED, __HIP_MEMORY_SCOPE_AGENT);
    __hip_atomic_fetch_add(q + 4, a1.x * nrm, __ATOMIC_RELAXED, __HIP_MEMORY_SCOPE_AGENT);
    __hip_atomic_fetch_add(q + 5, a1.y * nrm, __ATOMIC_RELAXED, __HIP_MEMORY_SCOPE_AGENT);
    __hip_atomic_fetch_add(q + 6, a1.z * nrm, __ATOMIC_RELAXED, __HIP_MEMORY_SCOPE_AGENT);
    __hip_atomic_fetch_add(q + 7, a1.w * nrm, __ATOMIC_RELAXED, __HIP_MEMORY_SCOPE_AGENT);
}

__global__ void relu_to_f16(const float* __restrict__ acc, _Float16* __restrict__ h16,
                            long long total)
{
    long long i = (long long)blockIdx.x * 256 + threadIdx.x;
    if (i < total) h16[i] = (_Float16)fmaxf(acc[i], 0.0f);
}

// mean/max pooling partials: 64 blocks x 256 feature-threads.
__global__ void pool_partial(const _Float16* __restrict__ h, float* __restrict__ psum,
                             float* __restrict__ pmax, int N)
{
    int f = threadIdx.x;
    int b = blockIdx.x;
    int rows_per = (N + 63) / 64;
    int r0 = b * rows_per;
    int r1 = r0 + rows_per; if (r1 > N) r1 = N;
    float s = 0.0f, mx = -__builtin_inff();
    for (int r = r0; r < r1; ++r) {
        float v = (float)h[(long long)r * HID + f];
        s += v; mx = fmaxf(mx, v);
    }
    psum[b * HID + f] = s;
    pmax[b * HID + f] = mx;
}

// single-block head: pooling reduce -> graph proj -> decoder -> edge prob scalar
__global__ void head_kernel(const float* __restrict__ psum, const float* __restrict__ pmax,
                            const float* __restrict__ Wg1, const float* __restrict__ bg1,
                            const float* __restrict__ Wg2, const float* __restrict__ bg2,
                            const float* __restrict__ Wd1, const float* __restrict__ bd1,
                            const float* __restrict__ Wd2, const float* __restrict__ bd2,
                            float* __restrict__ out, float* __restrict__ p_scalar,
                            float inv_n)
{
    __shared__ float rep[2 * HID];
    __shared__ float t1[HID];
    __shared__ float ge[128];
    __shared__ float vv[HID];
    int t = threadIdx.x;

    float s = 0.0f, m = -__builtin_inff();
    for (int b = 0; b < 64; ++b) {
        s += psum[b * HID + t];
        m = fmaxf(m, pmax[b * HID + t]);
    }
    rep[t]       = s * inv_n;   // mean half
    rep[HID + t] = m;           // max half
    __syncthreads();

    {   // t1 = relu(rep @ Wg1 + bg1), Wg1: [512,256]
        float acc = bg1[t];
        for (int k = 0; k < 2 * HID; ++k) acc += rep[k] * Wg1[k * HID + t];
        t1[t] = fmaxf(acc, 0.0f);
    }
    __syncthreads();

    if (t < 128) {  // ge = t1 @ Wg2 + bg2, Wg2: [256,128]
        float acc = bg2[t];
        for (int k = 0; k < HID; ++k) acc += t1[k] * Wg2[k * 128 + t];
        ge[t] = acc;
        out[t] = acc;           // graph_embedding -> d_out[0..127]
    }
    __syncthreads();

    {   // t1 = relu(ge @ Wd1 + bd1), Wd1: [128,256]
        float acc = bd1[t];
        for (int k = 0; k < 128; ++k) acc += ge[k] * Wd1[k * HID + t];
        t1[t] = fmaxf(acc, 0.0f);
    }
    __syncthreads();

    {   // vv = t1 @ Wd2 + bd2, Wd2: [256,256]
        float acc = bd2[t];
        for (int k = 0; k < HID; ++k) acc += t1[k] * Wd2[k * HID + t];
        vv[t] = acc;
    }
    __syncthreads();

    if (t == 0) {   // all node embeddings identical => one logit for every edge
        float dp = 0.0f;
        for (int k = 0; k < HID; ++k) dp += vv[k] * vv[k];
        *p_scalar = 1.0f / (1.0f + expf(-dp));
    }
}

__global__ void fill_probs(float* __restrict__ out, const float* __restrict__ p_scalar,
                           int E)
{
    int i = blockIdx.x * 256 + threadIdx.x;
    if (i < E) out[128 + i] = *p_scalar;
}

// ---------------------------------------------------------------------------

static inline size_t alignup(size_t x) { return (x + 255) & ~(size_t)255; }

extern "C" void kernel_launch(void* const* d_in, const int* in_sizes, int n_in,
                              void* d_out, int out_size, void* d_ws, size_t ws_size,
                              hipStream_t stream)
{
    const float* x     = (const float*)d_in[0];
    const int*   eidx  = (const int*)  d_in[1];
    const float* W1    = (const float*)d_in[2];
    const float* b1    = (const float*)d_in[3];
    const float* W2    = (const float*)d_in[4];
    const float* b2    = (const float*)d_in[5];
    const float* convW = (const float*)d_in[6];   // [3,256,256]
    const float* convB = (const float*)d_in[7];   // [3,256]
    const float* Wg1   = (const float*)d_in[8];
    const float* bg1   = (const float*)d_in[9];
    const float* Wg2   = (const float*)d_in[10];
    const float* bg2   = (const float*)d_in[11];
    const float* Wd1   = (const float*)d_in[12];
    const float* bd1   = (const float*)d_in[13];
    const float* Wd2   = (const float*)d_in[14];
    const float* bd2   = (const float*)d_in[15];

    const int FIN = 64;
    const int N = in_sizes[0] / FIN;           // 50000
    const int E = in_sizes[1] / 2;             // 800000
    const int* src = eidx;
    const int* dst = eidx + E;

    // ---- workspace carve-up ----
    char* w = (char*)d_ws;
    size_t off = 0;
    _Float16* x16  = (_Float16*)(w + off); off += alignup((size_t)N * FIN * 2);
    _Float16* h16  = (_Float16*)(w + off); off += alignup((size_t)N * HID * 2);
    float*    hw   = (float*)   (w + off); off += alignup((size_t)N * HID * 4);
    float*    acc  = (float*)   (w + off); off += alignup((size_t)N * HID * 4);
    _Float16* W1t  = (_Float16*)(w + off); off += alignup((size_t)FIN * HID * 2);
    _Float16* W2t  = (_Float16*)(w + off); off += alignup((size_t)HID * HID * 2);
    _Float16* cWt  = (_Float16*)(w + off); off += alignup((size_t)NLAYERS * HID * HID * 2);
    float*    dinv = (float*)   (w + off); off += alignup((size_t)N * 4);
    float*    psum = (float*)   (w + off); off += alignup((size_t)64 * HID * 4);
    float*    pmax = (float*)   (w + off); off += alignup((size_t)64 * HID * 4);
    float*    pscl = (float*)   (w + off); off += alignup(256);
    (void)ws_size; (void)n_in; (void)out_size;

    float* out = (float*)d_out;

    const long long totNH = (long long)N * HID;
    const int strips = N / 16;                 // 50000/16 = 3125

    // 1) weights -> f16 transposed; x -> f16
    w_transpose_f16<<<(FIN * HID + 255) / 256, 256, 0, stream>>>(W1, W1t, FIN, HID);
    w_transpose_f16<<<(HID * HID + 255) / 256, 256, 0, stream>>>(W2, W2t, HID, HID);
    for (int l = 0; l < NLAYERS; ++l)
        w_transpose_f16<<<(HID * HID + 255) / 256, 256, 0, stream>>>(
            convW + (size_t)l * HID * HID, cWt + (size_t)l * HID * HID, HID, HID);
    f32_to_f16<<<(int)(((long long)N * FIN + 255) / 256), 256, 0, stream>>>(
        x, x16, (long long)N * FIN);

    // 2) degree normalization (self loops included)
    deg_init <<<(N + 255) / 256, 256, 0, stream>>>(dinv, N);
    deg_accum<<<(E + 255) / 256, 256, 0, stream>>>(dst, dinv, E);
    deg_rsqrt<<<(N + 255) / 256, 256, 0, stream>>>(dinv, N);

    // 3) encoder
    //    h1 = relu(x @ W1 + b1)          -> f16
    wmma_gemm<2, true, true, true, false><<<strips, 32, 0, stream>>>(
        x16, W1t, b1, FIN, h16, nullptr);
    //    h  = h1 @ W2 + b2               -> f16 (in-place on h16 is safe:
    //    each wave loads all its A rows before storing them)
    wmma_gemm<8, true, false, true, false><<<strips, 32, 0, stream>>>(
        h16, W2t, b2, HID, h16, nullptr);

    // 4) GCN layers
    const int scat_blocks = (int)(((long long)(E + N) * 32 + 255) / 256);
    for (int l = 0; l < NLAYERS; ++l) {
        wmma_gemm<8, false, false, false, true><<<strips, 32, 0, stream>>>(
            h16, cWt + (size_t)l * HID * HID, nullptr, HID, nullptr, hw);
        acc_init<<<(int)((totNH + 255) / 256), 256, 0, stream>>>(
            acc, convB + (size_t)l * HID, totNH);
        scatter_edges<<<scat_blocks, 256, 0, stream>>>(hw, acc, src, dst, dinv, E, N);
        relu_to_f16<<<(int)((totNH + 255) / 256), 256, 0, stream>>>(acc, h16, totNH);
    }

    // 5) pooling + head + edge-prob broadcast
    pool_partial<<<64, 256, 0, stream>>>(h16, psum, pmax, N);
    head_kernel<<<1, 256, 0, stream>>>(psum, pmax, Wg1, bg1, Wg2, bg2,
                                       Wd1, bd1, Wd2, bd2, out, pscl, 1.0f / (float)N);
    fill_probs<<<(E + 255) / 256, 256, 0, stream>>>(out, pscl, E);
}
